// energy_cap_loss_61375082659859
// MI455X (gfx1250) — matrix operand order repaired and checked by reading, not compiled, not run on hardware
//
#include <hip/hip_runtime.h>
#include <hip/hip_bf16.h>

typedef __attribute__((ext_vector_type(16))) _Float16 v16h;
typedef __attribute__((ext_vector_type(8)))  float    v8f;

#define BATCH 4
#define H 256
#define W 256
#define HW (H*W)              // 65536
#define HN 254
#define NPIX (HN*HN)          // 64516
#define PTOT (BATCH*NPIX)     // 258064
#define CHUNK 128
#define MT 96                 // 81 padded to 6x16
#define GFR 60
#define POOLR 7

__device__ __constant__ float kSIGMA  = 0.041337f;
__device__ __constant__ float kGFEPS  = 0.001f;
__device__ __constant__ float kEPS    = 1e-5f;

// ---------------------------------------------------------------- elementwise
__global__ void depth_gray_kernel(const float* __restrict__ img,
                                  const float* __restrict__ noise,
                                  float* __restrict__ depth,
                                  float* __restrict__ gray) {
  int idx = blockIdx.x * blockDim.x + threadIdx.x;
  if (idx >= BATCH * HW) return;
  int b = idx >> 16;
  int p = idx & (HW - 1);
  float r  = img[(b*3 + 0)*HW + p];
  float g  = img[(b*3 + 1)*HW + p];
  float bl = img[(b*3 + 2)*HW + p];
  float rc = fminf(fmaxf(r,  -1e-4f), 1.0f);
  float gc = fminf(fmaxf(g,  -1e-4f), 1.0f);
  float bc = fminf(fmaxf(bl, -1e-4f), 1.0f);
  float mx = fmaxf(rc, fmaxf(gc, bc));
  float mn = fminf(rc, fminf(gc, bc));
  float safe = (mx == 0.0f) ? 1.0f : mx;
  float s = (mx == 0.0f) ? 0.0f : (mx - mn) / safe;
  depth[idx] = 0.121779f + 0.95971f * mx - 0.780245f * s + kSIGMA * noise[idx];
  gray[idx]  = 0.299f * r + 0.587f * g + 0.114f * bl;
}

// ---------------------------------------------------------------- min pool 15x15 (separable, clamped)
__global__ void minpool_w_kernel(const float* __restrict__ in, float* __restrict__ out) {
  int idx = blockIdx.x * blockDim.x + threadIdx.x;
  if (idx >= BATCH * HW) return;
  int b = idx >> 16, p = idx & (HW - 1);
  int i = p >> 8, j = p & 255;
  int j0 = max(j - POOLR, 0), j1 = min(j + POOLR, W - 1);
  const float* row = in + b*HW + i*W;
  float m = row[j0];
  for (int jj = j0 + 1; jj <= j1; ++jj) m = fminf(m, row[jj]);
  out[idx] = m;
}

__global__ void minpool_h_kernel(const float* __restrict__ in, float* __restrict__ out) {
  int idx = blockIdx.x * blockDim.x + threadIdx.x;
  if (idx >= BATCH * HW) return;
  int b = idx >> 16, p = idx & (HW - 1);
  int i = p >> 8, j = p & 255;
  int i0 = max(i - POOLR, 0), i1 = min(i + POOLR, H - 1);
  const float* col = in + b*HW + j;
  float m = col[i0 * W];
  for (int ii = i0 + 1; ii <= i1; ++ii) m = fminf(m, col[ii * W]);
  out[idx] = m;
}

// ---------------------------------------------------------------- clamped box filter (integral-image form)
__global__ void box_w_kernel(const float* __restrict__ in, float* __restrict__ cs,
                             float* __restrict__ out) {
  int row = blockIdx.x * blockDim.x + threadIdx.x;   // (b, i)
  if (row >= BATCH * H) return;
  const float* pi = in + (size_t)row * W;
  float* pc = cs + (size_t)row * W;
  float s = 0.0f;
  for (int j = 0; j < W; ++j) { s += pi[j]; pc[j] = s; }
  float* po = out + (size_t)row * W;
  for (int j = 0; j < W; ++j) {
    float hiv = pc[min(j + GFR, W - 1)];
    float lov = (j - GFR - 1 >= 0) ? pc[j - GFR - 1] : 0.0f;
    po[j] = hiv - lov;
  }
}

__global__ void box_h_kernel(const float* __restrict__ in, float* __restrict__ cs,
                             float* __restrict__ out) {
  int col = blockIdx.x * blockDim.x + threadIdx.x;   // (b, j)
  if (col >= BATCH * W) return;
  int b = col / W, j = col - b * W;
  const float* pi = in + (size_t)b*HW + j;
  float* pc = cs + (size_t)b*HW + j;
  float s = 0.0f;
  for (int i = 0; i < H; ++i) { s += pi[i*W]; pc[i*W] = s; }
  float* po = out + (size_t)b*HW + j;
  for (int i = 0; i < H; ++i) {
    float hiv = pc[min(i + GFR, H - 1) * W];
    float lov = (i - GFR - 1 >= 0) ? pc[(i - GFR - 1) * W] : 0.0f;
    po[i*W] = hiv - lov;
  }
}

__global__ void prod_kernel(const float* __restrict__ gray, const float* __restrict__ dR,
                            float* __restrict__ pxy, float* __restrict__ pxx) {
  int idx = blockIdx.x * blockDim.x + threadIdx.x;
  if (idx >= BATCH * HW) return;
  float g = gray[idx];
  pxy[idx] = g * dR[idx];
  pxx[idx] = g * g;
}

__device__ __forceinline__ float nbox_count(int i, int j) {
  int ci = min(i + GFR, H - 1) - max(i - GFR, 0) + 1;
  int cj = min(j + GFR, W - 1) - max(j - GFR, 0) + 1;
  return (float)(ci * cj);
}

__global__ void ab_kernel(const float* __restrict__ bx, const float* __restrict__ by,
                          const float* __restrict__ bxy, const float* __restrict__ bxx,
                          float* __restrict__ A, float* __restrict__ Bv) {
  int idx = blockIdx.x * blockDim.x + threadIdx.x;
  if (idx >= BATCH * HW) return;
  int p = idx & (HW - 1);
  int i = p >> 8, j = p & 255;
  float Nn = nbox_count(i, j);
  float inv = 1.0f / Nn;
  float mx = bx[idx] * inv, my = by[idx] * inv;
  float cov = bxy[idx] * inv - mx * my;
  float var = bxx[idx] * inv - mx * mx;
  float a = cov / (var + kGFEPS);
  A[idx]  = a;
  Bv[idx] = my - a * mx;
}

// ---------------------------------------------------------------- data term reduction
__global__ void data_term_kernel(const float* __restrict__ boxA, const float* __restrict__ boxb,
                                 const float* __restrict__ gray, const float* __restrict__ ypred,
                                 float* __restrict__ acc) {
  __shared__ float red[256];
  int idx = blockIdx.x * blockDim.x + threadIdx.x;
  float v = 0.0f;
  if (idx < BATCH * HW) {
    int p = idx & (HW - 1);
    int i = p >> 8, j = p & 255;
    float inv = 1.0f / nbox_count(i, j);
    float refined = boxA[idx] * inv * gray[idx] + boxb[idx] * inv;
    float t = fminf(fmaxf(expf(-refined), 0.05f), 1.0f);
    float d = ypred[idx] - t;
    v = d * d;
  }
  red[threadIdx.x] = v;
  __syncthreads();
  for (int s = 128; s > 0; s >>= 1) {
    if (threadIdx.x < s) red[threadIdx.x] += red[threadIdx.x + s];
    __syncthreads();
  }
  if (threadIdx.x == 0) atomicAdd(&acc[1], red[0]);
}

// ---------------------------------------------------------------- WMMA fragment loads from LDS
// A (16x32 f16, MxK): lanes 0-15 M=0..15, lanes 16-31 same M; VGPR0-3 hold K={hi*8+2g,+1},
// VGPR4-7 hold K=16+{hi*8+2g,+1}  (ISA 7.12.2)
__device__ __forceinline__ v16h load_a_frag(const _Float16* nd, int tm, int k0, int lane) {
  int m  = tm * 16 + (lane & 15);
  int hi = (lane >> 4) << 3;               // 0 or 8
  const _Float16* row = nd + m * CHUNK + k0;
  v16h a;
#pragma unroll
  for (int g = 0; g < 4; ++g) {
    a[2*g]     = row[hi + 2*g];
    a[2*g + 1] = row[hi + 2*g + 1];
  }
#pragma unroll
  for (int g = 0; g < 4; ++g) {
    a[8 + 2*g] = row[16 + hi + 2*g];
    a[9 + 2*g] = row[16 + hi + 2*g + 1];
  }
  return a;
}

// B (32x16 f16, KxN): lanes 0-15 hold K=0..7,16..23, lanes 16-31 hold K=8..15,24..31; N = lane&15.
// wt is stored transposed: wt[e][k] with e the N index, so K is contiguous per lane.
__device__ __forceinline__ v16h load_b_frag(const _Float16* wt, int tn, int k0, int lane) {
  int n  = tn * 16 + (lane & 15);
  int hi = (lane >> 4) << 3;               // 0 or 8
  const _Float16* row = wt + n * CHUNK + k0 + hi;
  v16h b;
#pragma unroll
  for (int h = 0; h < 16; ++h) b[h] = row[h < 8 ? h : h + 8];
  return b;
}

// ---------------------------------------------------------------- smoothness term (LDS-resident WMMA GEMM)
__global__ void __launch_bounds__(128)
smoothness_kernel(const float* __restrict__ ypred, const float* __restrict__ img,
                  float* __restrict__ acc) {
  __shared__ _Float16 ndL[MT * CHUNK];   // ND, rows = i (81 padded to 96), cols = pixel
  __shared__ _Float16 wTL[MT * CHUNK];   // W transposed, rows = e, cols = pixel
  __shared__ float red[CHUNK];

  const int t    = threadIdx.x;
  const int lane = t & 31;
  const int wv   = t >> 5;
  const float k9 = 1.0f / 9.0f;

  v8f zero8 = {0.f,0.f,0.f,0.f,0.f,0.f,0.f,0.f};
  v8f accs[9];
#pragma unroll
  for (int q = 0; q < 9; ++q) accs[q] = zero8;

  const int nchunks = (PTOT + CHUNK - 1) / CHUNK;
  for (int ch = blockIdx.x; ch < nchunks; ch += gridDim.x) {
    const int g = ch * CHUNK + t;
    if (g < PTOT) {
      const int b = g / NPIX;
      const int n = g - b * NPIX;
      const int i = n / HN;
      const int j = n - i * HN;

      const float* yb = ypred + (size_t)b * HW + i * W + j;
      float yp[9];
#pragma unroll
      for (int di = 0; di < 3; ++di)
#pragma unroll
        for (int dj = 0; dj < 3; ++dj)
          yp[di*3 + dj] = yb[di*W + dj];

      float xc[3][9];
#pragma unroll
      for (int c = 0; c < 3; ++c) {
        const float* ib = img + (size_t)(b*3 + c) * HW + i * W + j;
#pragma unroll
        for (int di = 0; di < 3; ++di)
#pragma unroll
          for (int dj = 0; dj < 3; ++dj)
            xc[c][di*3 + dj] = ib[di*W + dj];
      }

      float mean0 = 0.f, mean1 = 0.f, mean2 = 0.f;
#pragma unroll
      for (int p = 0; p < 9; ++p) { mean0 += xc[0][p]; mean1 += xc[1][p]; mean2 += xc[2][p]; }
      mean0 *= k9; mean1 *= k9; mean2 *= k9;

      float s00=0.f,s01=0.f,s02=0.f,s11=0.f,s12=0.f,s22=0.f;
#pragma unroll
      for (int p = 0; p < 9; ++p) {
        s00 += xc[0][p]*xc[0][p]; s01 += xc[0][p]*xc[1][p]; s02 += xc[0][p]*xc[2][p];
        s11 += xc[1][p]*xc[1][p]; s12 += xc[1][p]*xc[2][p]; s22 += xc[2][p]*xc[2][p];
      }
      // M = eps/9 * I + (XX^T/9 - mean mean^T)
      float m00 = s00*k9 - mean0*mean0 + kEPS*k9;
      float m01 = s01*k9 - mean0*mean1;
      float m02 = s02*k9 - mean0*mean2;
      float m11 = s11*k9 - mean1*mean1 + kEPS*k9;
      float m12 = s12*k9 - mean1*mean2;
      float m22 = s22*k9 - mean2*mean2 + kEPS*k9;
      // symmetric 3x3 inverse (adjugate / det)
      float c00 = m11*m22 - m12*m12;
      float c01 = m02*m12 - m01*m22;
      float c02 = m01*m12 - m02*m11;
      float det = m00*c00 + m01*c01 + m02*c02;
      float id  = 1.0f / det;
      float v00 = c00*id, v01 = c01*id, v02 = c02*id;
      float v11 = (m00*m22 - m02*m02)*id;
      float v12 = (m01*m02 - m00*m12)*id;
      float v22 = (m00*m11 - m01*m01)*id;

      float cc0[9], cc1[9], cc2[9], dd0[9], dd1[9], dd2[9];
#pragma unroll
      for (int p = 0; p < 9; ++p) {
        float a0 = xc[0][p]-mean0, a1 = xc[1][p]-mean1, a2 = xc[2][p]-mean2;
        cc0[p]=a0; cc1[p]=a1; cc2[p]=a2;
        dd0[p] = v00*a0 + v01*a1 + v02*a2;
        dd1[p] = v01*a0 + v11*a1 + v12*a2;
        dd2[p] = v02*a0 + v12*a1 + v22*a2;
      }
#pragma unroll
      for (int p = 0; p < 9; ++p) {
#pragma unroll
        for (int q = 0; q < 9; ++q) {
          int e = p*9 + q;                     // nd row i=e: (yp[i%9]-yp[i/9])^2 ; weight col e
          float dv = yp[q] - yp[p];
          ndL[e*CHUNK + t] = (_Float16)(dv * dv);
          float wvx = (cc0[p]*dd0[q] + cc1[p]*dd1[q] + cc2[p]*dd2[q] + 1.0f) * k9;
          wTL[e*CHUNK + t] = (_Float16)wvx;
        }
      }
    } else {
#pragma unroll
      for (int e = 0; e < 81; ++e) {
        ndL[e*CHUNK + t] = (_Float16)0.0f;
        wTL[e*CHUNK + t] = (_Float16)0.0f;
      }
    }
#pragma unroll
    for (int e = 81; e < MT; ++e) {            // zero pad rows 81..95
      ndL[e*CHUNK + t] = (_Float16)0.0f;
      wTL[e*CHUNK + t] = (_Float16)0.0f;
    }
    __syncthreads();

    // 36 C-tiles (6x6 of 16x16), 9 per wave, K swept in 4 steps of 32
#pragma unroll
    for (int q = 0; q < 9; ++q) {
      const int tid_ = wv + 4*q;
      const int tm = tid_ / 6;
      const int tn = tid_ - 6*tm;
#pragma unroll
      for (int k0 = 0; k0 < CHUNK; k0 += 32) {
        v16h av = load_a_frag(ndL, tm, k0, lane);
        v16h bv = load_b_frag(wTL, tn, k0, lane);
        accs[q] = __builtin_amdgcn_wmma_f32_16x16x32_f16(
            false, av, false, bv, (short)0, accs[q], false, false);
      }
    }
    __syncthreads();
  }

  // sum of ALL C entries == smoothness contribution; layout-agnostic reduce
  float part = 0.0f;
#pragma unroll
  for (int q = 0; q < 9; ++q)
#pragma unroll
    for (int r = 0; r < 8; ++r) part += accs[q][r];
  red[t] = part;
  __syncthreads();
  for (int s = 64; s > 0; s >>= 1) {
    if (t < s) red[t] += red[t + s];
    __syncthreads();
  }
  if (t == 0) atomicAdd(&acc[0], red[0]);
}

// ---------------------------------------------------------------- misc
__global__ void zero_acc_kernel(float* acc) { acc[0] = 0.0f; acc[1] = 0.0f; }

__global__ void finalize_kernel(const float* __restrict__ acc, float* __restrict__ out) {
  out[0] = (acc[0] + 0.01f * acc[1]) / (float)NPIX;
}

// ---------------------------------------------------------------- launch
extern "C" void kernel_launch(void* const* d_in, const int* in_sizes, int n_in,
                              void* d_out, int out_size, void* d_ws, size_t ws_size,
                              hipStream_t stream) {
  const float* img   = (const float*)d_in[0];   // (4,3,256,256)
  const float* ypred = (const float*)d_in[1];   // (4,1,256,256)
  const float* noise = (const float*)d_in[2];   // (4,1,256,256)
  float* out = (float*)d_out;

  float* w = (float*)d_ws;
  const size_t S = (size_t)BATCH * HW;          // 262144 floats per plane
  float* depth = w + 0*S;
  float* gray  = w + 1*S;
  float* cs    = w + 2*S;
  float* rtmp  = w + 3*S;
  float* pxy   = w + 4*S;
  float* pxx   = w + 5*S;
  float* bx    = w + 6*S;
  float* by    = w + 7*S;
  float* bxy   = w + 8*S;
  float* bxx   = w + 9*S;
  float* Aa    = w + 10*S;
  float* Bb    = w + 11*S;
  float* bA    = w + 12*S;
  float* bB    = w + 13*S;
  float* dR    = w + 14*S;
  float* accv  = w + 15*S;                      // [0]=smoothness, [1]=data

  dim3 ew(256), ewg(1024);                      // 262144 threads
  dim3 bf(256), bfg(4);                         // 1024 rows/cols

  hipLaunchKernelGGL(zero_acc_kernel, dim3(1), dim3(1), 0, stream, accv);
  hipLaunchKernelGGL(depth_gray_kernel, ewg, ew, 0, stream, img, noise, depth, gray);
  hipLaunchKernelGGL(minpool_w_kernel, ewg, ew, 0, stream, depth, rtmp);
  hipLaunchKernelGGL(minpool_h_kernel, ewg, ew, 0, stream, rtmp, dR);

  // boxfilter(gray) -> bx
  hipLaunchKernelGGL(box_w_kernel, bfg, bf, 0, stream, gray, cs, rtmp);
  hipLaunchKernelGGL(box_h_kernel, bfg, bf, 0, stream, rtmp, cs, bx);
  // boxfilter(depth_R) -> by
  hipLaunchKernelGGL(box_w_kernel, bfg, bf, 0, stream, dR, cs, rtmp);
  hipLaunchKernelGGL(box_h_kernel, bfg, bf, 0, stream, rtmp, cs, by);
  // products
  hipLaunchKernelGGL(prod_kernel, ewg, ew, 0, stream, gray, dR, pxy, pxx);
  hipLaunchKernelGGL(box_w_kernel, bfg, bf, 0, stream, pxy, cs, rtmp);
  hipLaunchKernelGGL(box_h_kernel, bfg, bf, 0, stream, rtmp, cs, bxy);
  hipLaunchKernelGGL(box_w_kernel, bfg, bf, 0, stream, pxx, cs, rtmp);
  hipLaunchKernelGGL(box_h_kernel, bfg, bf, 0, stream, rtmp, cs, bxx);
  // A, b then their box filters
  hipLaunchKernelGGL(ab_kernel, ewg, ew, 0, stream, bx, by, bxy, bxx, Aa, Bb);
  hipLaunchKernelGGL(box_w_kernel, bfg, bf, 0, stream, Aa, cs, rtmp);
  hipLaunchKernelGGL(box_h_kernel, bfg, bf, 0, stream, rtmp, cs, bA);
  hipLaunchKernelGGL(box_w_kernel, bfg, bf, 0, stream, Bb, cs, rtmp);
  hipLaunchKernelGGL(box_h_kernel, bfg, bf, 0, stream, rtmp, cs, bB);

  hipLaunchKernelGGL(data_term_kernel, ewg, ew, 0, stream, bA, bB, gray, ypred, accv);
  hipLaunchKernelGGL(smoothness_kernel, dim3(512), dim3(128), 0, stream, ypred, img, accv);
  hipLaunchKernelGGL(finalize_kernel, dim3(1), dim3(1), 0, stream, accv, out);
}
